// PotentialNet_35132832481482
// MI455X (gfx1250) — matrix operand order repaired and checked by reading, not compiled, not run on hardware
//
#include <hip/hip_runtime.h>
#include <hip/hip_bf16.h>

// ---------------- constants ----------------
#define NRES 384
#define CS   256
#define CP   128
#define CMUL 128
#define HH   12
#define CHID 16
#define PQK  4
#define PV   8
#define NBLK 4
#define MT   2          // m-subtiles per wave (32 rows)
#define NT   4          // n-subtiles per wave (64 cols)

static constexpr long long NN  = (long long)NRES * NRES;   // 147456
static constexpr long long NNC = NN * CP;                  // 18874368

typedef __attribute__((ext_vector_type(16))) __bf16 v16bf;
typedef __attribute__((ext_vector_type(8)))  float  v8f;

__device__ __forceinline__ __bf16 f2bf(float f) {
  union { float f; unsigned u; } x; x.f = f;
  unsigned r = x.u + 0x7FFFu + ((x.u >> 16) & 1u);   // RNE
  union { unsigned short s; __bf16 b; } y; y.s = (unsigned short)(r >> 16);
  return y.b;
}
__device__ __forceinline__ float bf2f(__bf16 b) {
  union { __bf16 b; unsigned short s; } x; x.b = b;
  union { unsigned u; float f; } y; y.u = ((unsigned)x.s) << 16;
  return y.f;
}
__device__ __forceinline__ float sigmoidf_(float v) { return 1.f / (1.f + __expf(-v)); }

// ================= fast WMMA GEMM (32x64 per wave) =================
// C[m][n] = alpha * sum_k A[m][k] * Wt[n][k] + bias[n]
// A: bf16 row-major, k contiguous, row stride lda (mult of 8)
// Wt: bf16 TRANSPOSED [Nout][K], k contiguous, row stride ldw (mult of 8)
// K must be a multiple of 32. epi: 0=none 1=relu 2=sigmoid. out_bf16: C dtype.
__global__ void gemm_bf16_fast_k(const __bf16* __restrict__ A, const __bf16* __restrict__ Wt,
                                 const float* __restrict__ bias, void* __restrict__ Cv,
                                 int M, int K, int Nout, int lda, int ldw, int ldc,
                                 float alpha, int epi, int out_bf16)
{
  int wave   = blockIdx.x * (blockDim.x >> 5) + (threadIdx.x >> 5);
  int tilesN = (Nout + 15) >> 4;
  int tilesM = (M + 15) >> 4;
  int tngs   = (tilesN + NT - 1) / NT;
  int tmgs   = (tilesM + MT - 1) / MT;
  if (wave >= tmgs * tngs) return;           // wave-uniform
  int tmg = wave / tngs, tng = wave % tngs;
  int lane = threadIdx.x & 31, hi = (lane >> 4) & 1, l16 = lane & 15;

  const __bf16* Ar[MT];
  const __bf16* Wr[NT];
  int mrow[MT], ncol[NT];
#pragma unroll
  for (int mt = 0; mt < MT; ++mt) {
    mrow[mt] = tmg * (16 * MT) + mt * 16 + l16;
    int mc = (mrow[mt] < M) ? mrow[mt] : (M - 1);
    Ar[mt] = A + (long long)mc * lda;
  }
#pragma unroll
  for (int nt = 0; nt < NT; ++nt) {
    ncol[nt] = tng * (16 * NT) + nt * 16 + l16;
    int nc = (ncol[nt] < Nout) ? ncol[nt] : (Nout - 1);
    Wr[nt] = Wt + (long long)nc * ldw;
  }

  v8f acc[MT][NT] = {};
  for (int k0 = 0; k0 < K; k0 += 32) {
    union U { uint4 u[2]; v16bf v; } af[MT], bf[NT];
#pragma unroll
    for (int mt = 0; mt < MT; ++mt) {
      af[mt].u[0] = *(const uint4*)(Ar[mt] + k0 + 8 * hi);
      af[mt].u[1] = *(const uint4*)(Ar[mt] + k0 + 16 + 8 * hi);
    }
#pragma unroll
    for (int nt = 0; nt < NT; ++nt) {
      bf[nt].u[0] = *(const uint4*)(Wr[nt] + k0 + 8 * hi);
      bf[nt].u[1] = *(const uint4*)(Wr[nt] + k0 + 16 + 8 * hi);
    }
#pragma unroll
    for (int mt = 0; mt < MT; ++mt)
#pragma unroll
      for (int nt = 0; nt < NT; ++nt)
        acc[mt][nt] = __builtin_amdgcn_wmma_f32_16x16x32_bf16(
            false, af[mt].v, false, bf[nt].v, (short)0, acc[mt][nt], false, false);
  }
#pragma unroll
  for (int nt = 0; nt < NT; ++nt) {
    int nc = (ncol[nt] < Nout) ? ncol[nt] : (Nout - 1);
    float bv = bias ? bias[nc] : 0.f;
#pragma unroll
    for (int mt = 0; mt < MT; ++mt) {
#pragma unroll
      for (int r = 0; r < 8; ++r) {
        int row = tmg * (16 * MT) + mt * 16 + (hi << 3) + r;
        if (row < M && ncol[nt] < Nout) {
          float v = acc[mt][nt][r] * alpha + bv;
          if (epi == 1)      v = v > 0.f ? v : 0.f;
          else if (epi == 2) v = sigmoidf_(v);
          long long o = (long long)row * ldc + ncol[nt];
          if (out_bf16) ((__bf16*)Cv)[o] = f2bf(v);
          else          ((float*)Cv)[o]  = v;
        }
      }
    }
  }
}

// ================= general-stride WMMA GEMM (small IPA GEMMs) =================
// C[b][m][n] = alpha * sum_k A[b][m][k] * W[b][(k,n)]
__global__ void gemm_bf16_gen_k(const __bf16* __restrict__ A, const __bf16* __restrict__ W,
                                float* __restrict__ C,
                                int M, int K, int Nout, int lda, int ldb, int ldbn, int ldc,
                                long long sA, long long sW, long long sC, float alpha)
{
  int wave   = blockIdx.x * (blockDim.x >> 5) + (threadIdx.x >> 5);
  int tilesN = (Nout + 15) >> 4;
  int tilesM = (M + 15) >> 4;
  if (wave >= tilesM * tilesN) return;
  long long b = blockIdx.y;
  A += b * sA; W += b * sW; C += b * sC;
  int tm = wave / tilesN, tn = wave % tilesN;
  int m0 = tm << 4, n0 = tn << 4;
  int lane = threadIdx.x & 31, hi = (lane >> 4) & 1, l16 = lane & 15;
  int mrow = m0 + l16; int mc = (mrow < M) ? mrow : (M - 1);
  int ncol = n0 + l16; int nc = (ncol < Nout) ? ncol : (Nout - 1);
  const __bf16* Ar = A + (long long)mc * lda;
  __bf16 zb = f2bf(0.f);

  v8f acc = {};
  for (int k0 = 0; k0 < K; k0 += 32) {
    v16bf Af, Bf;
#pragma unroll
    for (int p = 0; p < 8; ++p) {
      int kk = (p < 4) ? (8 * hi + 2 * p) : (16 + 8 * hi + 2 * (p - 4));
      int k  = k0 + kk;
      Af[2 * p]     = (k     < K) ? Ar[k]     : zb;
      Af[2 * p + 1] = (k + 1 < K) ? Ar[k + 1] : zb;
    }
#pragma unroll
    for (int e = 0; e < 16; ++e) {
      int k = k0 + ((e < 8) ? (8 * hi + e) : (16 + 8 * hi + (e - 8)));
      Bf[e] = (k < K) ? W[(long long)k * ldb + (long long)nc * ldbn] : zb;
    }
    acc = __builtin_amdgcn_wmma_f32_16x16x32_bf16(false, Af, false, Bf, (short)0, acc,
                                                  false, false);
  }
#pragma unroll
  for (int r = 0; r < 8; ++r) {
    int row = m0 + (hi << 3) + r;
    if (row < M && ncol < Nout)
      C[(long long)row * ldc + ncol] = acc[r] * alpha;
  }
}

// ================= triangle einsum (32x64 per wave) =================
// X[i,j,c] = sum_k At[c][i][k] * Bt[c][j][k]
__global__ void tri_einsum_k(const __bf16* __restrict__ At, const __bf16* __restrict__ Bt,
                             float* __restrict__ X)
{
  const int Nn = NRES;
  int wave = blockIdx.x * (blockDim.x >> 5) + (threadIdx.x >> 5);
  int tmgs = (Nn / 16) / MT;   // 12
  int tngs = (Nn / 16) / NT;   // 6
  if (wave >= tmgs * tngs) return;
  int c = blockIdx.y;
  const __bf16* Ac = At + (long long)c * NN;
  const __bf16* Bc = Bt + (long long)c * NN;
  int tmg = wave / tngs, tng = wave % tngs;
  int lane = threadIdx.x & 31, hi = (lane >> 4) & 1, l16 = lane & 15;
  const __bf16* Ar[MT];
  const __bf16* Br[NT];
#pragma unroll
  for (int mt = 0; mt < MT; ++mt)
    Ar[mt] = Ac + (long long)(tmg * (16 * MT) + mt * 16 + l16) * Nn;
#pragma unroll
  for (int nt = 0; nt < NT; ++nt)
    Br[nt] = Bc + (long long)(tng * (16 * NT) + nt * 16 + l16) * Nn;

  v8f acc[MT][NT] = {};
  for (int k0 = 0; k0 < Nn; k0 += 32) {
    union U { uint4 u[2]; v16bf v; } af[MT], bf[NT];
#pragma unroll
    for (int mt = 0; mt < MT; ++mt) {
      af[mt].u[0] = *(const uint4*)(Ar[mt] + k0 + 8 * hi);
      af[mt].u[1] = *(const uint4*)(Ar[mt] + k0 + 16 + 8 * hi);
    }
#pragma unroll
    for (int nt = 0; nt < NT; ++nt) {
      bf[nt].u[0] = *(const uint4*)(Br[nt] + k0 + 8 * hi);
      bf[nt].u[1] = *(const uint4*)(Br[nt] + k0 + 16 + 8 * hi);
    }
#pragma unroll
    for (int mt = 0; mt < MT; ++mt)
#pragma unroll
      for (int nt = 0; nt < NT; ++nt)
        acc[mt][nt] = __builtin_amdgcn_wmma_f32_16x16x32_bf16(
            false, af[mt].v, false, bf[nt].v, (short)0, acc[mt][nt], false, false);
  }
#pragma unroll
  for (int mt = 0; mt < MT; ++mt)
#pragma unroll
    for (int nt = 0; nt < NT; ++nt)
#pragma unroll
      for (int r = 0; r < 8; ++r) {
        int row = tmg * (16 * MT) + mt * 16 + (hi << 3) + r;
        int col = tng * (16 * NT) + nt * 16 + l16;
        X[((long long)row * Nn + col) * CP + c] = acc[mt][nt][r];
      }
}

// ================= layernorm (wave per row, optional fp32 + bf16 outs) =======
__global__ void layernorm_k(const float* __restrict__ x, const float* __restrict__ g,
                            const float* __restrict__ b, float* __restrict__ yf,
                            __bf16* __restrict__ yb, long long rows, int dim)
{
  long long row = (long long)blockIdx.x * (blockDim.x >> 5) + (threadIdx.x >> 5);
  if (row >= rows) return;
  int lane = threadIdx.x & 31;
  const float* xr = x + row * dim;
  float s = 0.f, s2 = 0.f;
  for (int i = lane; i < dim; i += 32) { float v = xr[i]; s += v; s2 += v * v; }
#pragma unroll
  for (int o = 16; o; o >>= 1) { s += __shfl_xor(s, o, 32); s2 += __shfl_xor(s2, o, 32); }
  float mu  = s / dim;
  float inv = rsqrtf(s2 / dim - mu * mu + 1e-5f);
  for (int i = lane; i < dim; i += 32) {
    float v = (xr[i] - mu) * inv * g[i] + b[i];
    if (yf) yf[row * dim + i] = v;
    if (yb) yb[row * dim + i] = f2bf(v);
  }
}

// ================= conversions / elementwise =================
__global__ void wt_cvt_k(const float* __restrict__ w, __bf16* __restrict__ wt, int K, int N) {
  int idx = blockIdx.x * blockDim.x + threadIdx.x;   // idx = n*K + k
  if (idx >= K * N) return;
  int k = idx % K, n = idx / K;
  wt[idx] = f2bf(w[(long long)k * N + n]);
}
__global__ void cvt_f2b_k(const float* __restrict__ s, __bf16* __restrict__ d, long long n) {
  long long i = (long long)blockIdx.x * blockDim.x + threadIdx.x;
  if (i < n) d[i] = f2bf(s[i]);
}
__global__ void scale_copy_k(float* dst, const float* src, long long n, float a) {
  long long i = (long long)blockIdx.x * blockDim.x + threadIdx.x;
  if (i < n) dst[i] = src[i] * a;
}
__global__ void add_k(float* dst, const float* a, const float* b, long long n) {
  long long i = (long long)blockIdx.x * blockDim.x + threadIdx.x;
  if (i < n) dst[i] = a[i] + b[i];
}
// a = mask * sigmoid(G) * P  -> channel-major bf16; transpose swaps (i1,i2)
__global__ void gate_pack_k(const __bf16* __restrict__ G, const __bf16* __restrict__ Pl,
                            const float* __restrict__ pm, __bf16* __restrict__ out_t,
                            int transpose) {
  long long idx = (long long)blockIdx.x * blockDim.x + threadIdx.x;
  if (idx >= NNC) return;
  long long ij = idx >> 7; int c = (int)(idx & 127);
  float v = pm[ij] * sigmoidf_(bf2f(G[idx])) * bf2f(Pl[idx]);
  long long i1 = ij / NRES, i2 = ij % NRES;
  long long dst = (long long)c * NN + (transpose ? (i2 * NRES + i1) : ij);
  out_t[dst] = f2bf(v);
}
__global__ void resid_gate_k(float* P, const float* Z, const float* G, long long n) {
  long long i = (long long)blockIdx.x * blockDim.x + threadIdx.x;
  if (i < n) P[i] += Z[i] * sigmoidf_(G[i]);
}
__global__ void pt_combine_k(float* P, const float* X, const float* pm, long long n) {
  long long i = (long long)blockIdx.x * blockDim.x + threadIdx.x;
  if (i < n) { float m = pm[i >> 7]; P[i] = (P[i] + X[i] * m) * m; }
}
// out = R * p + t  (fp32 out; optional bf16 copy)
__global__ void pts_transform_k(const float* __restrict__ ptl, const float* __restrict__ rot,
                                const float* __restrict__ trn, float* __restrict__ out,
                                __bf16* __restrict__ outb, int npts) {
  int idx = blockIdx.x * blockDim.x + threadIdx.x;
  if (idx >= NRES * npts) return;
  int n = idx / npts;
  const float* R = rot + n * 9;
  const float* p = ptl + (long long)idx * 3;
  float x = p[0], y = p[1], z = p[2];
  float o0 = R[0] * x + R[1] * y + R[2] * z + trn[n * 3 + 0];
  float o1 = R[3] * x + R[4] * y + R[5] * z + trn[n * 3 + 1];
  float o2 = R[6] * x + R[7] * y + R[8] * z + trn[n * 3 + 2];
  float* o = out + (long long)idx * 3;
  o[0] = o0; o[1] = o1; o[2] = o2;
  if (outb) { outb[idx * 3 + 0] = f2bf(o0); outb[idx * 3 + 1] = f2bf(o1); outb[idx * 3 + 2] = f2bf(o2); }
}

// ================= attention finalize + softmax (block per (h,i)) ============
__global__ void attn_softmax_k(const float* __restrict__ LOG, __bf16* __restrict__ ATTb,
                               const float* __restrict__ BZ, const float* __restrict__ QPG,
                               const float* __restrict__ KVPG, const float* __restrict__ mask,
                               const float* __restrict__ head_w)
{
  const int Nn = NRES;
  int h = blockIdx.x / Nn, i = blockIdx.x % Nn;
  int j = threadIdx.x;
  __shared__ float red[12];
  __shared__ float bc;
  float l = LOG[((long long)h * Nn + i) * Nn + j];
  l += 0.57735026919f * BZ[((long long)i * Nn + j) * HH + h];
  float d2 = 0.f;
#pragma unroll
  for (int p = 0; p < PQK; ++p) {
    const float* qp = QPG + ((long long)i * (HH * PQK) + h * PQK + p) * 3;
    const float* kp = KVPG + ((long long)j * (HH * (PQK + PV)) + h * (PQK + PV) + p) * 3;
    float dx = qp[0] - kp[0], dy = qp[1] - kp[1], dz = qp[2] - kp[2];
    d2 += dx * dx + dy * dy + dz * dz;
  }
  float w  = head_w[h];
  float hw = (w > 20.f) ? w : log1pf(__expf(w));
  l += d2 * hw * (-0.5f * 0.13608276348f);
  l += (mask[i] * mask[j] - 1.f) * 100000.f;
  float m = l;
#pragma unroll
  for (int o = 16; o; o >>= 1) m = fmaxf(m, __shfl_xor(m, o, 32));
  if ((threadIdx.x & 31) == 0) red[threadIdx.x >> 5] = m;
  __syncthreads();
  if (threadIdx.x < 32) {
    float t = (threadIdx.x < 12) ? red[threadIdx.x] : -1e30f;
#pragma unroll
    for (int o = 16; o; o >>= 1) t = fmaxf(t, __shfl_xor(t, o, 32));
    if (threadIdx.x == 0) bc = t;
  }
  __syncthreads();
  m = bc;
  float e = __expf(l - m);
  float s = e;
#pragma unroll
  for (int o = 16; o; o >>= 1) s += __shfl_xor(s, o, 32);
  if ((threadIdx.x & 31) == 0) red[threadIdx.x >> 5] = s;
  __syncthreads();
  if (threadIdx.x < 32) {
    float t = (threadIdx.x < 12) ? red[threadIdx.x] : 0.f;
#pragma unroll
    for (int o = 16; o; o >>= 1) t += __shfl_xor(t, o, 32);
    if (threadIdx.x == 0) bc = t;
  }
  __syncthreads();
  ATTb[((long long)h * Nn + i) * Nn + j] = f2bf(e / bc);
}

// ================= o_pt local frame + norms -> cat[192..575] =================
__global__ void opt_finalize_k(const float* __restrict__ OPT, const float* __restrict__ rot,
                               const float* __restrict__ trn, float* __restrict__ cat)
{
  int idx = blockIdx.x * blockDim.x + threadIdx.x;
  if (idx >= NRES * HH * PV) return;
  int n = idx / (HH * PV), hp = idx % (HH * PV);
  const float* R = rot + n * 9;
  const float* p = OPT + (long long)idx * 3;
  float x = p[0] - trn[n * 3 + 0], y = p[1] - trn[n * 3 + 1], z = p[2] - trn[n * 3 + 2];
  float lx = R[0] * x + R[3] * y + R[6] * z;
  float ly = R[1] * x + R[4] * y + R[7] * z;
  float lz = R[2] * x + R[5] * y + R[8] * z;
  float nr = sqrtf(lx * lx + ly * ly + lz * lz + 1e-8f);
  float* c = cat + (long long)n * 2112;
  c[192 + hp] = lx;  c[288 + hp] = ly;  c[384 + hp] = lz;  c[480 + hp] = nr;
}

// ================= backbone compose =================
__global__ void bb_update_k(const float* __restrict__ U6, const float* __restrict__ mask,
                            float* rot, float* trn)
{
  int n = blockIdx.x * blockDim.x + threadIdx.x;
  if (n >= NRES) return;
  float m  = mask[n];
  float q1 = U6[n * 6 + 0] * m, q2 = U6[n * 6 + 1] * m, q3 = U6[n * 6 + 2] * m;
  float t0 = U6[n * 6 + 3] * m, t1 = U6[n * 6 + 4] * m, t2 = U6[n * 6 + 5] * m;
  float inv = rsqrtf(1.f + q1 * q1 + q2 * q2 + q3 * q3);
  float a = inv, b = q1 * inv, c = q2 * inv, d = q3 * inv;
  float R[9] = { a*a + b*b - c*c - d*d, 2*(b*c - a*d),         2*(b*d + a*c),
                 2*(b*c + a*d),         a*a - b*b + c*c - d*d, 2*(c*d - a*b),
                 2*(b*d - a*c),         2*(c*d + a*b),         a*a - b*b - c*c + d*d };
  float* Ro = rot + n * 9;
  float old[9];
#pragma unroll
  for (int i = 0; i < 9; ++i) old[i] = Ro[i];
#pragma unroll
  for (int x = 0; x < 3; ++x)
#pragma unroll
    for (int z = 0; z < 3; ++z) {
      float s = 0.f;
#pragma unroll
      for (int y = 0; y < 3; ++y) s += old[x * 3 + y] * R[y * 3 + z];
      Ro[x * 3 + z] = s;
    }
  float* T = trn + n * 3;
  float tt0 = T[0], tt1 = T[1], tt2 = T[2];
  T[0] = tt0 + old[0] * t0 + old[1] * t1 + old[2] * t2;
  T[1] = tt1 + old[3] * t0 + old[4] * t1 + old[5] * t2;
  T[2] = tt2 + old[6] * t0 + old[7] * t1 + old[8] * t2;
}

__global__ void finalize_out_k(const float* S, const float* R, const float* T, float* out) {
  int i = blockIdx.x * blockDim.x + threadIdx.x;
  if (i < NRES * CS) out[i] = S[i];
  if (i < NRES * 9)  out[NRES * CS + i] = R[i];
  if (i < NRES * 3)  out[NRES * CS + NRES * 9 + i] = T[i] * 10.f;
}

// ---------------- param flattening (jax pytree: dict keys sorted) ------------
enum {
  BB_B = 0, BB_W, IPA_BIAS_B, IPA_BIAS_W, IPA_HEADW, IPA_KV_B, IPA_KV_W,
  IPA_KVP_B, IPA_KVP_W, IPA_OUT_B, IPA_OUT_W, IPA_Q_B, IPA_Q_W,
  IPA_QP_B, IPA_QP_W, IPALN_B, IPALN_G,
  PT_L1_B, PT_L1_W, PT_L2_B, PT_L2_W, PT_LN_B, PT_LN_G,
  TR_L1_B, TR_L1_W, TR_L2_B, TR_L2_W, TR_L3_B, TR_L3_W,
  TRLN_B, TRLN_G,
  TRI_IN = 31, TRI_OUT = 47, PER_BLK = 63
};
enum { T_AG_B = 0, T_AG_W, T_AP_B, T_AP_W, T_BG_B, T_BG_W, T_BP_B, T_BP_W,
       T_G_B, T_G_W, T_LNI_B, T_LNI_G, T_LNO_B, T_LNO_G, T_Z_B, T_Z_W };

// fp32 small-region offsets
static constexpr long long OFF_S    = 0;
static constexpr long long OFF_S2   = 98304;
static constexpr long long OFF_SU   = 196608;
static constexpr long long OFF_ROT  = 294912;
static constexpr long long OFF_TRN  = 298368;
static constexpr long long OFF_QPL  = 299520;
static constexpr long long OFF_QPG  = 354816;
static constexpr long long OFF_KVPL = 410112;
static constexpr long long OFF_KVPG = 576000;
static constexpr long long OFF_BZ   = 741888;
static constexpr long long OFF_LOG  = 2511360;
static constexpr long long OFF_OPT  = 4280832;
static constexpr long long OFF_CAT  = 4391424;
static constexpr long long OFF_U6   = 5202432;
static constexpr long long FP32_SMALL = 5204736;
// bf16 small-region offsets
static constexpr long long BO_SB    = 0;
static constexpr long long BO_QB    = 98304;
static constexpr long long BO_KVB   = 172032;
static constexpr long long BO_KVPGB = 319488;
static constexpr long long BO_ATTB  = 485376;
static constexpr long long BO_CATB  = 2254848;
static constexpr long long BO_H1B   = 3065856;
static constexpr long long BO_H2B   = 3164160;
static constexpr long long BF16_SMALL = 3262464;

static inline int ew_blocks(long long n) { return (int)((n + 255) / 256); }

extern "C" void kernel_launch(void* const* d_in, const int* in_sizes, int n_in,
                              void* d_out, int out_size, void* d_ws, size_t ws_size,
                              hipStream_t stream)
{
  (void)in_sizes; (void)n_in; (void)out_size; (void)ws_size;
  const float* s_in     = (const float*)d_in[0];
  const float* p_in     = (const float*)d_in[1];
  const float* rot_in   = (const float*)d_in[2];
  const float* trans_in = (const float*)d_in[3];
  const float* mask     = (const float*)d_in[4];
  const float* pmask    = (const float*)d_in[5];
  auto prm = [&](int blk, int idx) -> const float* {
    return (const float*)d_in[6 + blk * PER_BLK + idx];
  };

  // ---- workspace layout ----
  float* F   = (float*)d_ws;
  float* P   = F;                 // NNC fp32
  float* X   = F + NNC;           // NNC fp32
  float* TG  = F + 2 * NNC;       // NNC fp32
  float* SMr = F + 3 * NNC;       // small fp32 region
  float* S   = SMr + OFF_S;    float* S2  = SMr + OFF_S2;   float* SU  = SMr + OFF_SU;
  float* ROT = SMr + OFF_ROT;  float* TRN = SMr + OFF_TRN;
  float* QPL = SMr + OFF_QPL;  float* QPG = SMr + OFF_QPG;
  float* KVPL= SMr + OFF_KVPL; float* KVPG= SMr + OFF_KVPG;
  float* BZ  = SMr + OFF_BZ;   float* LOG = SMr + OFF_LOG;
  float* OPT = SMr + OFF_OPT;  float* CAT = SMr + OFF_CAT;  float* U6 = SMr + OFF_U6;

  __bf16* Bb   = (__bf16*)(SMr + FP32_SMALL);
  __bf16* ZLNb = Bb;               // NNC
  __bf16* G1b  = Bb + 1 * NNC;     // NNC (aliases: XLNb, Hb start)
  __bf16* G2b  = Bb + 2 * NNC;     // NNC (alias: Pb)
  __bf16* ATb  = Bb + 3 * NNC;     // NNC
  __bf16* BTb  = Bb + 4 * NNC;     // NNC
  __bf16* Hb   = G1b;              // NN x 512 spans G1b..BTb (pair-transition phase)
  __bf16* XLNb = G1b;
  __bf16* Pb   = G2b;
  __bf16* SB   = Bb + 5 * NNC;     // small bf16 region
  __bf16* Sb    = SB + BO_SB;    __bf16* Qb    = SB + BO_QB;   __bf16* KVb = SB + BO_KVB;
  __bf16* KVPGb = SB + BO_KVPGB; __bf16* ATTb  = SB + BO_ATTB;
  __bf16* CATb  = SB + BO_CATB;  __bf16* H1b   = SB + BO_H1B;  __bf16* H2b = SB + BO_H2B;
  __bf16* WT    = SB + BF16_SMALL;   // transposed bf16 weight arena

  // ---- convert all weights (transposed [N][K], bf16) ----
  long long woff[NBLK][PER_BLK] = {};
  {
    const int wl[][3] = {
      {BB_W,256,6},{IPA_BIAS_W,128,12},{IPA_KV_W,256,384},{IPA_KVP_W,256,432},
      {IPA_OUT_W,2112,256},{IPA_Q_W,256,192},{IPA_QP_W,256,144},
      {PT_L1_W,128,512},{PT_L2_W,512,128},
      {TR_L1_W,256,256},{TR_L2_W,256,256},{TR_L3_W,256,256},
      {TRI_IN+T_AG_W,128,128},{TRI_IN+T_AP_W,128,128},{TRI_IN+T_BG_W,128,128},
      {TRI_IN+T_BP_W,128,128},{TRI_IN+T_G_W,128,128},{TRI_IN+T_Z_W,128,128},
      {TRI_OUT+T_AG_W,128,128},{TRI_OUT+T_AP_W,128,128},{TRI_OUT+T_BG_W,128,128},
      {TRI_OUT+T_BP_W,128,128},{TRI_OUT+T_G_W,128,128},{TRI_OUT+T_Z_W,128,128},
    };
    long long cur = 0;
    for (int blk = 0; blk < NBLK; ++blk)
      for (unsigned e = 0; e < sizeof(wl)/sizeof(wl[0]); ++e) {
        int idx = wl[e][0], K = wl[e][1], N = wl[e][2];
        woff[blk][idx] = cur;
        wt_cvt_k<<<ew_blocks((long long)K * N), 256, 0, stream>>>(prm(blk, idx), WT + cur, K, N);
        cur += (long long)K * N;
        cur = (cur + 7) & ~7LL;
      }
  }
  auto wt = [&](int blk, int idx) -> const __bf16* { return WT + woff[blk][idx]; };

  auto gemmF = [&](const __bf16* A, const __bf16* Wt_, const float* bias, void* C,
                   long long M, int K, int Nout, int lda, int ldc, float alpha,
                   int epi, int obf) {
    int tmgs = (int)(((M + 15) / 16 + MT - 1) / MT);
    int tngs = (((Nout + 15) / 16) + NT - 1) / NT;
    int waves = tmgs * tngs;
    gemm_bf16_fast_k<<<dim3((waves + 7) / 8), 256, 0, stream>>>(
        A, Wt_, bias, C, (int)M, K, Nout, lda, K, ldc, alpha, epi, obf);
  };
  auto gemmG = [&](const __bf16* A, const __bf16* W, float* C,
                   int M, int K, int Nout, int lda, int ldb, int ldbn, int ldc,
                   long long sA, long long sW, long long sC, int nb, float alpha) {
    int tiles = ((M + 15) / 16) * ((Nout + 15) / 16);
    gemm_bf16_gen_k<<<dim3((tiles + 7) / 8, nb), 256, 0, stream>>>(
        A, W, C, M, K, Nout, lda, ldb, ldbn, ldc, sA, sW, sC, alpha);
  };
  auto lnorm = [&](const float* x, const float* g, const float* b, float* yf, __bf16* yb,
                   long long rows, int dim) {
    layernorm_k<<<(unsigned)((rows + 7) / 8), 256, 0, stream>>>(x, g, b, yf, yb, rows, dim);
  };

  // ---- init ----
  hipMemcpyAsync(P, p_in, NNC * sizeof(float), hipMemcpyDeviceToDevice, stream);
  hipMemcpyAsync(S, s_in, NRES * CS * sizeof(float), hipMemcpyDeviceToDevice, stream);
  hipMemcpyAsync(ROT, rot_in, NRES * 9 * sizeof(float), hipMemcpyDeviceToDevice, stream);
  cvt_f2b_k<<<ew_blocks(NRES * CS), 256, 0, stream>>>(s_in, Sb, NRES * CS);
  scale_copy_k<<<ew_blocks(NRES * 3), 256, 0, stream>>>(TRN, trans_in, NRES * 3, 0.1f);

  auto tri_mul = [&](int blk, int tb, int outgoing) {
    auto q = [&](int i) { return prm(blk, tb + i); };
    auto w = [&](int i) { return wt(blk, tb + i); };
    lnorm(P, q(T_LNI_G), q(T_LNI_B), nullptr, ZLNb, NN, CP);
    gemmF(ZLNb, w(T_AG_W), q(T_AG_B), G1b, NN, CP, CMUL, CP, CMUL, 1.f, 0, 1);
    gemmF(ZLNb, w(T_AP_W), q(T_AP_B), G2b, NN, CP, CMUL, CP, CMUL, 1.f, 0, 1);
    gate_pack_k<<<ew_blocks(NNC), 256, 0, stream>>>(G1b, G2b, pmask, ATb, !outgoing);
    gemmF(ZLNb, w(T_BG_W), q(T_BG_B), G1b, NN, CP, CMUL, CP, CMUL, 1.f, 0, 1);
    gemmF(ZLNb, w(T_BP_W), q(T_BP_B), G2b, NN, CP, CMUL, CP, CMUL, 1.f, 0, 1);
    gate_pack_k<<<ew_blocks(NNC), 256, 0, stream>>>(G1b, G2b, pmask, BTb, !outgoing);
    {
      int waves = ((NRES / 16) / MT) * ((NRES / 16) / NT);   // 72
      tri_einsum_k<<<dim3((waves + 7) / 8, CMUL), 256, 0, stream>>>(ATb, BTb, X);
    }
    lnorm(X, q(T_LNO_G), q(T_LNO_B), nullptr, XLNb, NN, CMUL);
    gemmF(XLNb, w(T_Z_W), q(T_Z_B), X, NN, CMUL, CP, CMUL, CP, 1.f, 0, 0);
    gemmF(ZLNb, w(T_G_W), q(T_G_B), TG, NN, CP, CP, CP, CP, 1.f, 0, 0);
    resid_gate_k<<<ew_blocks(NNC), 256, 0, stream>>>(P, X, TG, NNC);
  };

  for (int blk = 0; blk < NBLK; ++blk) {
    // --- pair track ---
    tri_mul(blk, TRI_OUT, 1);
    tri_mul(blk, TRI_IN, 0);
    // pair transition (bf16 hidden aliases the 4 gate/einsum temps)
    lnorm(P, prm(blk, PT_LN_G), prm(blk, PT_LN_B), nullptr, ZLNb, NN, CP);
    gemmF(ZLNb, wt(blk, PT_L1_W), prm(blk, PT_L1_B), Hb, NN, CP, 4 * CP, CP, 4 * CP, 1.f, 1, 1);
    gemmF(Hb, wt(blk, PT_L2_W), prm(blk, PT_L2_B), X, NN, 4 * CP, CP, 4 * CP, CP, 1.f, 0, 0);
    pt_combine_k<<<ew_blocks(NNC), 256, 0, stream>>>(P, X, pmask, NNC);

    // --- IPA ---
    cvt_f2b_k<<<ew_blocks(NNC), 256, 0, stream>>>(P, Pb, NNC);
    gemmF(Sb, wt(blk, IPA_Q_W),   prm(blk, IPA_Q_B),   Qb,   NRES, CS, HH * CHID, CS, HH * CHID, 1.f, 0, 1);
    gemmF(Sb, wt(blk, IPA_KV_W),  prm(blk, IPA_KV_B),  KVb,  NRES, CS, 2 * HH * CHID, CS, 2 * HH * CHID, 1.f, 0, 1);
    gemmF(Sb, wt(blk, IPA_QP_W),  prm(blk, IPA_QP_B),  QPL,  NRES, CS, HH * PQK * 3, CS, HH * PQK * 3, 1.f, 0, 0);
    gemmF(Sb, wt(blk, IPA_KVP_W), prm(blk, IPA_KVP_B), KVPL, NRES, CS, HH * (PQK + PV) * 3, CS, HH * (PQK + PV) * 3, 1.f, 0, 0);
    pts_transform_k<<<ew_blocks(NRES * HH * PQK), 256, 0, stream>>>(QPL, ROT, TRN, QPG, nullptr, HH * PQK);
    pts_transform_k<<<ew_blocks(NRES * HH * (PQK + PV)), 256, 0, stream>>>(KVPL, ROT, TRN, KVPG, KVPGb, HH * (PQK + PV));
    gemmF(Pb, wt(blk, IPA_BIAS_W), prm(blk, IPA_BIAS_B), BZ, NN, CP, HH, CP, HH, 1.f, 0, 0);
    // logits (K=16 zero-padded), batched over heads
    gemmG(Qb, KVb, LOG, NRES, CHID, NRES,
          HH * CHID, 1, 2 * HH * CHID, NRES, CHID, 2 * CHID, NN, HH, 0.14433756730f);
    attn_softmax_k<<<HH * NRES, NRES, 0, stream>>>(LOG, ATTb, BZ, QPG, KVPG, mask,
                                                   prm(blk, IPA_HEADW));
    // o = attn @ v -> cat[0:192]
    gemmG(ATTb, KVb + CHID, CAT, NRES, NRES, CHID,
          NRES, 2 * HH * CHID, 1, 2112, NN, 2 * CHID, CHID, HH, 1.f);
    // o_pt(global) = attn @ v_pts
    gemmG(ATTb, KVPGb + PQK * 3, OPT, NRES, NRES, PV * 3,
          NRES, HH * (PQK + PV) * 3, 1, HH * PV * 3, NN, (PQK + PV) * 3, PV * 3, HH, 1.f);
    // o_pair = attn @ z, batched over residues -> cat[576:2112]
    gemmG(ATTb, Pb, CAT + 576, HH, NRES, CP,
          (int)NN, CP, 1, CP, NRES, (long long)NRES * CP, 2112, NRES, 1.f);
    opt_finalize_k<<<ew_blocks(NRES * HH * PV), 256, 0, stream>>>(OPT, ROT, TRN, CAT);
    cvt_f2b_k<<<ew_blocks(NRES * 2112), 256, 0, stream>>>(CAT, CATb, NRES * 2112);
    gemmF(CATb, wt(blk, IPA_OUT_W), prm(blk, IPA_OUT_B), SU, NRES, 2112, CS, 2112, CS, 1.f, 0, 0);
    add_k<<<ew_blocks(NRES * CS), 256, 0, stream>>>(S2, S, SU, NRES * CS);
    lnorm(S2, prm(blk, IPALN_G), prm(blk, IPALN_B), S, Sb, NRES, CS);

    // --- transition ---
    gemmF(Sb,  wt(blk, TR_L1_W), prm(blk, TR_L1_B), H1b, NRES, CS, CS, CS, CS, 1.f, 1, 1);
    gemmF(H1b, wt(blk, TR_L2_W), prm(blk, TR_L2_B), H2b, NRES, CS, CS, CS, CS, 1.f, 1, 1);
    gemmF(H2b, wt(blk, TR_L3_W), prm(blk, TR_L3_B), SU,  NRES, CS, CS, CS, CS, 1.f, 0, 0);
    add_k<<<ew_blocks(NRES * CS), 256, 0, stream>>>(S2, S, SU, NRES * CS);
    lnorm(S2, prm(blk, TRLN_G), prm(blk, TRLN_B), S, Sb, NRES, CS);

    // --- backbone compose ---
    gemmF(Sb, wt(blk, BB_W), prm(blk, BB_B), U6, NRES, CS, 6, CS, 6, 1.f, 0, 0);
    bb_update_k<<<ew_blocks(NRES), 256, 0, stream>>>(U6, mask, ROT, TRN);
  }

  finalize_out_k<<<ew_blocks(NRES * CS), 256, 0, stream>>>(S, ROT, TRN, (float*)d_out);
}
